// BiLSTM_50551765074576
// MI455X (gfx1250) — compile-verified, matrix operand order checked by reading, not run on hardware
//
#include <hip/hip_runtime.h>

// Problem constants (from reference)
#define T_TOK 16384   // total tokens = 128 segments * 128
#define D_INP 1024
#define HID   512
#define G4    2048    // 4*H gates per direction
#define LSEG  128
#define SPLIT 4       // workgroups cooperating per (batch-tile, dir) group

typedef __attribute__((ext_vector_type(16))) __bf16 v16bf;
typedef __attribute__((ext_vector_type(8)))  __bf16 v8bf;
typedef __attribute__((ext_vector_type(8)))  float  v8f;

// Per-lane 16-element bf16 WMMA fragment: elements {0..7} at p, {8..15} at
// p+16 — matches ISA 7.12.2 16-bit A/B layouts when p = row_base + k0 +
// ((lane&16)?8:0).
__device__ __forceinline__ v16bf frag16(const __bf16* p) {
    v8bf lo = *(const v8bf*)p;
    v8bf hi = *(const v8bf*)(p + 16);
    return __builtin_shufflevector(lo, hi, 0,1,2,3,4,5,6,7,8,9,10,11,12,13,14,15);
}

__device__ __forceinline__ float sigf(float x) { return 1.0f / (1.0f + __expf(-x)); }

// ---------------- async global->LDS copy (ASYNCcnt path) ----------------
#if __has_builtin(__builtin_amdgcn_global_load_async_to_lds_b128)
#define HAS_ASYNC_LDS 1
#else
#define HAS_ASYNC_LDS 0
#endif

// builtin expects int4 (gcc vector_size) pointers: AS1 source, AS3 dest
typedef int v4i_vs __attribute__((vector_size(4 * sizeof(int))));
typedef __attribute__((address_space(1))) v4i_vs gbl_v4i;
typedef __attribute__((address_space(3))) v4i_vs lds_v4i;

// copy 8 bf16 (16B) global -> LDS
__device__ __forceinline__ void async_cp16(const __bf16* g, __bf16* l) {
#if HAS_ASYNC_LDS
    __builtin_amdgcn_global_load_async_to_lds_b128(
        (gbl_v4i*)(uintptr_t)g, (lds_v4i*)(uintptr_t)l, 0, 0);
#else
    *(v8bf*)l = *(const v8bf*)g;
#endif
}

template <int N>
__device__ __forceinline__ void async_wait() {
#if HAS_ASYNC_LDS && __has_builtin(__builtin_amdgcn_s_wait_asynccnt)
    __builtin_amdgcn_s_wait_asynccnt(N);
#endif
}

// ---------------- cross-WG group barrier (monotonic counter) -------------
// Release: flush this thread's stores to L2 (agent scope). Acquire after the
// spin: invalidate the read-only WGP$ so next-step h reads are not stale.
__device__ __forceinline__ void grp_barrier(unsigned* ctr, unsigned target) {
    __builtin_amdgcn_fence(__ATOMIC_RELEASE, "agent");
    __syncthreads();
    if (threadIdx.x == 0) {
        __hip_atomic_fetch_add(ctr, 1u, __ATOMIC_RELAXED, __HIP_MEMORY_SCOPE_AGENT);
        while (__hip_atomic_load(ctr, __ATOMIC_RELAXED, __HIP_MEMORY_SCOPE_AGENT) < target)
            __builtin_amdgcn_s_sleep(2);
    }
    __syncthreads();
    __builtin_amdgcn_fence(__ATOMIC_ACQUIRE, "agent");
}

// ---------------- f32 -> bf16 conversion ----------------
__global__ void f2bf(const float* __restrict__ src, __bf16* __restrict__ dst, int n) {
    for (int i = blockIdx.x * blockDim.x + threadIdx.x; i < n; i += gridDim.x * blockDim.x)
        dst[i] = (__bf16)src[i];
}

__global__ void zero_u32(unsigned* p, int n) {
    int i = blockIdx.x * blockDim.x + threadIdx.x;
    if (i < n) p[i] = 0u;
}

// ---------------- input projection: XP = X * W_ih^T + bias ----------------
// 128x128 macro-tile per block (256 threads = 8 waves), K chunks of 32 staged
// in LDS with async double-buffering. Wave w owns the 16-row strip (w*16) and
// all 8 N sub-tiles -> 8 WMMAs per A-fragment. B-fragments are batch-loaded
// ahead of the WMMA chain so DScnt partial waits overlap loads with compute.
__global__ __launch_bounds__(256) void gemm_xp(const __bf16* __restrict__ X,
                                               const __bf16* __restrict__ W,
                                               const float*  __restrict__ bias,
                                               float* __restrict__ XP) {
    __shared__ __attribute__((aligned(16))) __bf16 Ab[2][128 * 32];
    __shared__ __attribute__((aligned(16))) __bf16 Bb[2][128 * 32];

    const int tid  = threadIdx.x;
    const int wave = tid >> 5;
    const int lane = tid & 31;
    const int lr   = lane & 15;
    const int koff = (lane & 16) ? 8 : 0;
    const int n0g  = blockIdx.x * 128;    // 2048/128 = 16
    const int m0   = blockIdx.y * 128;    // 16384/128 = 128

    const __bf16* Xrow = X + (size_t)m0  * D_INP;
    const __bf16* Wrow = W + (size_t)n0g * D_INP;

    v8f acc[8];
#pragma unroll
    for (int n = 0; n < 8; ++n) {
        const float bv = bias[n0g + n * 16 + lr];
#pragma unroll
        for (int r = 0; r < 8; ++r) acc[n][r] = bv;
    }

    // stage K-chunk k0 into buffer sel: 512 16B units per matrix, 2 per thread
    auto stage = [&](int k0, int sel) {
#pragma unroll
        for (int u = tid; u < 512; u += 256) {
            const int row = u >> 2, cu = (u & 3) * 8;
            async_cp16(Xrow + (size_t)row * D_INP + k0 + cu, &Ab[sel][row * 32 + cu]);
            async_cp16(Wrow + (size_t)row * D_INP + k0 + cu, &Bb[sel][row * 32 + cu]);
        }
    };

    stage(0, 0);
    for (int i = 0; i < D_INP / 32; ++i) {
        const int cur = i & 1;
        if ((i + 1) < (D_INP / 32)) {
            stage((i + 1) * 32, cur ^ 1);  // 4 more asyncs in flight
            async_wait<4>();               // retire chunk i (in-order completion)
        } else {
            async_wait<0>();
        }
        __syncthreads();

        // issue all fragment loads first, then the WMMA chain
        v16bf a = frag16(&Ab[cur][(wave * 16 + lr) * 32 + koff]);
        v16bf bfr[8];
#pragma unroll
        for (int n = 0; n < 8; ++n)
            bfr[n] = frag16(&Bb[cur][(n * 16 + lr) * 32 + koff]);
#pragma unroll
        for (int n = 0; n < 8; ++n)
            acc[n] = __builtin_amdgcn_wmma_f32_16x16x32_bf16(false, a, false, bfr[n],
                                                             (short)0, acc[n],
                                                             false, false);
        __syncthreads();   // all reads done before buf is overwritten (chunk i+2)
    }

    const int rbase = m0 + wave * 16 + ((lane & 16) ? 8 : 0);
    float* out = XP + n0g + lr;
#pragma unroll
    for (int n = 0; n < 8; ++n)
#pragma unroll
        for (int r = 0; r < 8; ++r)
            out[(size_t)(rbase + r) * G4 + n * 16] = acc[n][r];
}

// ---------------- recurrence (gate-split across SPLIT WGs per group) ------
// grid: (SPLIT, 8 batch tiles, 2 dirs); block 256 = 8 waves.
// Group (btile,dir): 4 WGs cooperate; WG q owns h columns [q*128,(q+1)*128).
// Wave w owns h cols hcol0 = q*128+w*16 and its 4 gate tiles (i/f/g/o at
// offsets {0,512,1024,1536}+hcol0 in the 2048-wide gate space). h is
// exchanged through a double-buffered 16x512 bf16 slab in L2, one group
// barrier per step.
__global__ __launch_bounds__(256) void lstm_rec(const float*  __restrict__ XPf,
                                                const float*  __restrict__ XPb,
                                                const __bf16* __restrict__ WHHf,
                                                const __bf16* __restrict__ WHHb,
                                                __bf16* __restrict__ Hbuf,
                                                unsigned* __restrict__ Ctr,
                                                float* __restrict__ OUT) {
    const int q     = blockIdx.x;   // gate-split slice
    const int btile = blockIdx.y;
    const int dir   = blockIdx.z;
    const int grp   = dir * 8 + btile;

    const float*  XP  = dir ? XPb  : XPf;
    const __bf16* WHH = dir ? WHHb : WHHf;
    __bf16* H0 = Hbuf + (size_t)grp * (2 * 16 * HID);
    __bf16* H1 = H0 + 16 * HID;
    unsigned* ctr = Ctr + grp;

    const int wave = threadIdx.x >> 5;
    const int lane = threadIdx.x & 31;
    const int lr   = lane & 15;
    const int koff = (lane & 16) ? 8 : 0;   // K-half select for A/B frags
    const int rsel = (lane & 16) ? 8 : 0;   // row-half select for C/D
    const int hcol0 = q * 128 + wave * 16;  // this wave's h columns
    unsigned target = SPLIT;

    // zero own 16x128 slice of H0 (the t=0 hidden state)
    for (int i = threadIdx.x; i < 16 * 128; i += 256) {
        const int rr = i >> 7, cc = i & 127;
        H0[rr * HID + q * 128 + cc] = (__bf16)0.0f;
    }
    grp_barrier(ctr, target);
    target += SPLIT;

    v8f c;
#pragma unroll
    for (int r = 0; r < 8; ++r) c[r] = 0.0f;

    for (int t = 0; t < LSEG; ++t) {
        const int tt = dir ? (LSEG - 1 - t) : t;
        const __bf16* Hc = (t & 1) ? H1 : H0;
        __bf16*       Hn = (t & 1) ? H0 : H1;

        // gate accumulators initialized from xp (+bias, folded in gemm)
        v8f g[4];
#pragma unroll
        for (int gi = 0; gi < 4; ++gi) {
            const int n0 = gi * HID + hcol0;
#pragma unroll
            for (int r = 0; r < 8; ++r) {
                const size_t tok = (size_t)((btile * 16 + rsel + r) * LSEG + tt);
                g[gi][r] = XP[tok * G4 + n0 + lr];
            }
        }

        // prefetch next step's xp tile (each row is its own cacheline)
        if (t + 1 < LSEG) {
            const int tn = dir ? (LSEG - 2 - t) : (t + 1);
#pragma unroll
            for (int gi = 0; gi < 4; ++gi) {
                const int n0 = gi * HID + hcol0;
#pragma unroll
                for (int r = 0; r < 8; ++r) {
                    const size_t tok = (size_t)((btile * 16 + rsel + r) * LSEG + tn);
                    __builtin_prefetch(&XP[tok * G4 + n0 + lr], 0, 0);
                }
            }
        }

        // g += h @ W_hh^T  (A from L2 h-slab, B = W_hh rows from L2);
        // fully unrolled so fragment loads of chunk k+1 overlap WMMAs of k
#pragma unroll
        for (int k0 = 0; k0 < HID; k0 += 32) {
            v16bf a = frag16(Hc + lr * HID + k0 + koff);
            v16bf bfr[4];
#pragma unroll
            for (int gi = 0; gi < 4; ++gi)
                bfr[gi] = frag16(WHH + (size_t)(gi * HID + hcol0 + lr) * HID + k0 + koff);
#pragma unroll
            for (int gi = 0; gi < 4; ++gi)
                g[gi] = __builtin_amdgcn_wmma_f32_16x16x32_bf16(false, a, false, bfr[gi],
                                                                (short)0, g[gi],
                                                                false, false);
        }

        // elementwise cell update (identical element mapping across the 4 tiles)
#pragma unroll
        for (int r = 0; r < 8; ++r) {
            const float iv = sigf(g[0][r]);
            const float fv = sigf(g[1][r]);
            const float gv = tanhf(g[2][r]);
            const float ov = sigf(g[3][r]);
            c[r] = fv * c[r] + iv * gv;
            const float hv = ov * tanhf(c[r]);
            const int m = rsel + r;
            Hn[m * HID + hcol0 + lr] = (__bf16)hv;   // next-step A operand
            const size_t tok = (size_t)((btile * 16 + m) * LSEG + tt);
            OUT[tok * 1024 + dir * HID + hcol0 + lr] = hv;
        }

        grp_barrier(ctr, target);   // writers done -> next step may read Hn
        target += SPLIT;
    }
}

__global__ void copy_bnd(const int* __restrict__ src, int* __restrict__ dst, int n) {
    int i = blockIdx.x * blockDim.x + threadIdx.x;
    if (i < n) dst[i] = src[i];
}

extern "C" void kernel_launch(void* const* d_in, const int* in_sizes, int n_in,
                              void* d_out, int out_size, void* d_ws, size_t ws_size,
                              hipStream_t stream) {
    const float* embs = (const float*)d_in[0];
    const int*   bnd  = (const int*)d_in[1];
    const float* Wihf = (const float*)d_in[2];
    const float* Whhf = (const float*)d_in[3];
    const float* bf   = (const float*)d_in[4];
    const float* Wihb = (const float*)d_in[5];
    const float* Whhb = (const float*)d_in[6];
    const float* bb   = (const float*)d_in[7];

    char* ws = (char*)d_ws;
    // bf16 staging: x 32MiB, W_ih 4MiB x2, W_hh 2MiB x2; xp f32 128MiB x2;
    // h exchange slabs + barrier counters after 300MiB
    __bf16* x_bf   = (__bf16*)(ws);
    __bf16* wihf_b = (__bf16*)(ws + ((size_t)32 << 20));
    __bf16* wihb_b = (__bf16*)(ws + ((size_t)36 << 20));
    __bf16* whhf_b = (__bf16*)(ws + ((size_t)40 << 20));
    __bf16* whhb_b = (__bf16*)(ws + ((size_t)42 << 20));
    float*  xpf    = (float*)(ws + ((size_t)44 << 20));
    float*  xpb    = (float*)(ws + ((size_t)172 << 20));
    __bf16* hbuf   = (__bf16*)(ws + ((size_t)300 << 20));  // 16 grp * 2 * 16*512 bf16 = 512KiB
    unsigned* ctr  = (unsigned*)(ws + ((size_t)301 << 20));

    f2bf<<<4096, 256, 0, stream>>>(embs, x_bf, T_TOK * D_INP);
    f2bf<<<2048, 256, 0, stream>>>(Wihf, wihf_b, G4 * D_INP);
    f2bf<<<2048, 256, 0, stream>>>(Wihb, wihb_b, G4 * D_INP);
    f2bf<<<1024, 256, 0, stream>>>(Whhf, whhf_b, G4 * HID);
    f2bf<<<1024, 256, 0, stream>>>(Whhb, whhb_b, G4 * HID);
    zero_u32<<<1, 32, 0, stream>>>(ctr, 16);

    // 16 N-macro x 128 M-macro blocks per direction
    gemm_xp<<<dim3(16, 128), 256, 0, stream>>>(x_bf, wihf_b, bf, xpf);
    gemm_xp<<<dim3(16, 128), 256, 0, stream>>>(x_bf, wihb_b, bb, xpb);

    lstm_rec<<<dim3(SPLIT, 8, 2), 256, 0, stream>>>(xpf, xpb, whhf_b, whhb_b,
                                                    hbuf, ctr, (float*)d_out);

    int nb = in_sizes[1];
    copy_bnd<<<(nb + 127) / 128, 128, 0, stream>>>(
        bnd, (int*)((float*)d_out + (size_t)T_TOK * 1024), nb);
}
